// MultiheadLatentAttentionTensorCast_39505109189253
// MI455X (gfx1250) — matrix operand order repaired and checked by reading, not compiled, not run on hardware
//
#include <hip/hip_runtime.h>

// ---------------- problem constants ----------------
#define Bb    2
#define Ss    1024
#define Hh    2048
#define NHh   16
#define NOPEd 128
#define ROPEd 64
#define VDd   128
#define QKDd  192        // NOPE + ROPE
#define QLRd  1536
#define KVLRd 512
#define QKC   576        // KVLR + ROPE (concat score dim)
#define Mrows (Bb*Ss)    // 2048
#define MASK_MIN (-3.0e38f)
#define APITCH 40        // LDS row pitch in bf16 elems (80B: 16B-aligned, conflict-free)
#define BPITCH 40

typedef __attribute__((ext_vector_type(16))) __bf16 bf16x16;
typedef __attribute__((ext_vector_type(8)))  __bf16 bf16x8;
typedef __attribute__((ext_vector_type(8)))  float  f32x8;
typedef __attribute__((ext_vector_type(4)))  int    v4i;

// float -> bf16, round-to-nearest-even
__device__ __forceinline__ __bf16 f2bf(float f) {
    unsigned int u = __builtin_bit_cast(unsigned int, f);
    unsigned int lsb = (u >> 16) & 1u;
    u += 0x7FFFu + lsb;
    unsigned short s = (unsigned short)(u >> 16);
    return __builtin_bit_cast(__bf16, s);
}

__device__ __forceinline__ f32x8 wmma_bf16(bf16x16 a, bf16x16 b, f32x8 c) {
    return __builtin_amdgcn_wmma_f32_16x16x32_bf16(
        false, a, false, b, (short)0, c, false, false);
}

// A fragment (row-major source): two contiguous 16B runs.
__device__ __forceinline__ bf16x16 load_a_frag(const __bf16* row, int kk, int half) {
    bf16x8 lo = *(const bf16x8*)(row + kk + half * 16);
    bf16x8 hi = *(const bf16x8*)(row + kk + half * 16 + 8);
    return __builtin_shufflevector(lo, hi, 0,1,2,3,4,5,6,7,8,9,10,11,12,13,14,15);
}
// B fragment from global ([N][K] row-major): one 32B run.
__device__ __forceinline__ bf16x16 load_b_frag(const __bf16* row, int kk, int half) {
    return *(const bf16x16*)(row + kk + half * 16);
}
// B fragment from LDS (80B row pitch => only 16B alignment): two 16B runs.
__device__ __forceinline__ bf16x16 load_b_frag_lds(const __bf16* row, int half) {
    bf16x8 lo = *(const bf16x8*)(row + half * 16);
    bf16x8 hi = *(const bf16x8*)(row + half * 16 + 8);
    return __builtin_shufflevector(lo, hi, 0,1,2,3,4,5,6,7,8,9,10,11,12,13,14,15);
}

// ---- CDNA5 async global->LDS copy (ASYNCcnt path), 16B per lane ----
typedef __attribute__((address_space(1))) v4i* g_v4i_ptr;
typedef __attribute__((address_space(3))) v4i* l_v4i_ptr;

__device__ __forceinline__ void async_copy16(__bf16* lds_dst, const __bf16* gsrc) {
#if __has_builtin(__builtin_amdgcn_global_load_async_to_lds_b128)
    __builtin_amdgcn_global_load_async_to_lds_b128(
        (g_v4i_ptr)(void*)gsrc, (l_v4i_ptr)(void*)lds_dst, 0, 0);
#else
    // VDST = LDS byte address (low 32 bits of flat shared addr), VADDR = global addr
    asm volatile("global_load_async_to_lds_b128 %0, %1, off"
                 :: "v"((unsigned int)(unsigned long long)lds_dst), "v"(gsrc)
                 : "memory");
#endif
}
__device__ __forceinline__ void wait_async_zero() {
#if __has_builtin(__builtin_amdgcn_s_wait_asynccnt)
    __builtin_amdgcn_s_wait_asynccnt(0);
#else
    asm volatile("s_wait_asynccnt 0" ::: "memory");
#endif
}

// =========================================================================================
// Workgroup-cooperative GEMM: 4 waves, C tile 64x128, LDS double-buffered async pipeline.
// D = A(MxK) * B(NxK)^T. gridDim = (N/128, M/64, Z), blockDim = 128.
// =========================================================================================
template<bool OUT_BF16>
__global__ void gemm_wmma_lds(const __bf16* __restrict__ A, int lda, long long aStrideZ,
                              const __bf16* __restrict__ B, int ldb, long long bStrideZ,
                              void* __restrict__ Cv, int ldc, long long cStrideZ,
                              int K)
{
    __shared__ alignas(16) __bf16 As[2][64 * APITCH];
    __shared__ alignas(16) __bf16 Bs[2][128 * BPITCH];

    const int tid  = threadIdx.x;
    const int lane = tid & 31;
    const int half = lane >> 4;
    const int l15  = lane & 15;
    const int w    = tid >> 5;
    const int wm   = (w >> 1) * 32;   // wave's M sub-offset (0 / 32)
    const int wn   = (w & 1) * 64;    // wave's N sub-offset (0 / 64)
    const int n0 = blockIdx.x * 128;
    const int m0 = blockIdx.y * 64;
    const int z  = blockIdx.z;

    const __bf16* Ag = A + (long long)z * aStrideZ;
    const __bf16* Bg = B + (long long)z * bStrideZ;

    auto fill = [&](int buf, int kk) {
        // A panel: 64 rows x 64B = 256 x 16B chunks, 2 per thread
        #pragma unroll
        for (int i = 0; i < 2; ++i) {
            const int c = tid + i * 128;
            const int row = c >> 2, part = c & 3;
            async_copy16(&As[buf][row * APITCH + part * 8],
                         Ag + (long long)(m0 + row) * lda + kk + part * 8);
        }
        // B panel: 128 rows x 64B = 512 x 16B chunks, 4 per thread
        #pragma unroll
        for (int i = 0; i < 4; ++i) {
            const int c = tid + i * 128;
            const int row = c >> 2, part = c & 3;
            async_copy16(&Bs[buf][row * BPITCH + part * 8],
                         Bg + (long long)(n0 + row) * ldb + kk + part * 8);
        }
    };

    fill(0, 0);
    f32x8 acc[2][4] = {};
    const int nsteps = K >> 5;
    for (int s = 0; s < nsteps; ++s) {
        wait_async_zero();        // own async fills of buf[s&1] complete
        __syncthreads();          // ...visible to all waves; all waves done with buf[(s+1)&1]
        if (s + 1 < nsteps) fill((s + 1) & 1, (s + 1) << 5);

        const __bf16* a0 = &As[s & 1][(wm + l15) * APITCH];
        const __bf16* a1 = a0 + 16 * APITCH;
        const __bf16* b0 = &Bs[s & 1][(wn + l15) * BPITCH];
        const __bf16* b1 = b0 + 16 * BPITCH;
        const __bf16* b2 = b0 + 32 * BPITCH;
        const __bf16* b3 = b0 + 48 * BPITCH;

        bf16x16 af0 = load_a_frag(a0, 0, half);
        bf16x16 af1 = load_a_frag(a1, 0, half);
        bf16x16 bf0 = load_b_frag_lds(b0, half);
        bf16x16 bf1 = load_b_frag_lds(b1, half);
        bf16x16 bf2 = load_b_frag_lds(b2, half);
        bf16x16 bf3 = load_b_frag_lds(b3, half);
        acc[0][0] = wmma_bf16(af0, bf0, acc[0][0]);
        acc[0][1] = wmma_bf16(af0, bf1, acc[0][1]);
        acc[0][2] = wmma_bf16(af0, bf2, acc[0][2]);
        acc[0][3] = wmma_bf16(af0, bf3, acc[0][3]);
        acc[1][0] = wmma_bf16(af1, bf0, acc[1][0]);
        acc[1][1] = wmma_bf16(af1, bf1, acc[1][1]);
        acc[1][2] = wmma_bf16(af1, bf2, acc[1][2]);
        acc[1][3] = wmma_bf16(af1, bf3, acc[1][3]);
    }
    #pragma unroll
    for (int mi = 0; mi < 2; ++mi)
      #pragma unroll
      for (int j = 0; j < 4; ++j)
        #pragma unroll
        for (int e = 0; e < 8; ++e) {
            const int r = m0 + wm + mi * 16 + half * 8 + e;
            const int c = n0 + wn + j * 16 + l15;
            const long long off = (long long)z * cStrideZ + (long long)r * ldc + c;
            if (OUT_BF16) ((__bf16*)Cv)[off] = f2bf(acc[mi][j][e]);
            else          ((float*)Cv)[off]  = acc[mi][j][e];
        }
}

// ---------------- single-wave direct-global GEMM (for N%128!=0 cases) ----------------
// gridDim = (N/64, M/32, Z); blockDim = 32
template<bool OUT_BF16>
__global__ void gemm_wmma(const __bf16* __restrict__ A, int lda, long long aStrideZ,
                          const __bf16* __restrict__ B, int ldb, long long bStrideZ,
                          void* __restrict__ Cv, int ldc, long long cStrideZ,
                          int K)
{
    const int lane = threadIdx.x & 31;
    const int half = lane >> 4;
    const int l15  = lane & 15;
    const int n0 = blockIdx.x * 64;
    const int m0 = blockIdx.y * 32;
    const int z  = blockIdx.z;

    const __bf16* a0 = A + (long long)z * aStrideZ + (long long)(m0 + l15) * lda;
    const __bf16* a1 = a0 + 16LL * lda;
    const __bf16* bbase = B + (long long)z * bStrideZ + (long long)(n0 + l15) * ldb;
    const __bf16* b0 = bbase;
    const __bf16* b1 = bbase + 16LL * ldb;
    const __bf16* b2 = bbase + 32LL * ldb;
    const __bf16* b3 = bbase + 48LL * ldb;

    f32x8 acc[2][4] = {};
    for (int kk = 0; kk < K; kk += 32) {
        bf16x16 af0 = load_a_frag(a0, kk, half);
        bf16x16 af1 = load_a_frag(a1, kk, half);
        bf16x16 bf0 = load_b_frag(b0, kk, half);
        bf16x16 bf1 = load_b_frag(b1, kk, half);
        bf16x16 bf2 = load_b_frag(b2, kk, half);
        bf16x16 bf3 = load_b_frag(b3, kk, half);
        acc[0][0] = wmma_bf16(af0, bf0, acc[0][0]);
        acc[0][1] = wmma_bf16(af0, bf1, acc[0][1]);
        acc[0][2] = wmma_bf16(af0, bf2, acc[0][2]);
        acc[0][3] = wmma_bf16(af0, bf3, acc[0][3]);
        acc[1][0] = wmma_bf16(af1, bf0, acc[1][0]);
        acc[1][1] = wmma_bf16(af1, bf1, acc[1][1]);
        acc[1][2] = wmma_bf16(af1, bf2, acc[1][2]);
        acc[1][3] = wmma_bf16(af1, bf3, acc[1][3]);
    }
    #pragma unroll
    for (int mi = 0; mi < 2; ++mi)
      #pragma unroll
      for (int j = 0; j < 4; ++j)
        #pragma unroll
        for (int e = 0; e < 8; ++e) {
            const int r = m0 + mi * 16 + half * 8 + e;
            const int c = n0 + j * 16 + l15;
            const long long off = (long long)z * cStrideZ + (long long)r * ldc + c;
            if (OUT_BF16) ((__bf16*)Cv)[off] = f2bf(acc[mi][j][e]);
            else          ((float*)Cv)[off]  = acc[mi][j][e];
        }
}

// ---------------- causal masked score GEMM: scores = qcat * kcat^T * scale ----------------
// gridDim = (S/64 key tiles, S/32 query tiles, B*NH); blockDim = 32
__global__ void attn_scores(const __bf16* __restrict__ qcat, const __bf16* __restrict__ kcat,
                            float* __restrict__ scores)
{
    const int lane = threadIdx.x & 31;
    const int half = lane >> 4;
    const int l15  = lane & 15;
    const int n0 = blockIdx.x * 64;   // key
    const int m0 = blockIdx.y * 32;   // query
    const int z  = blockIdx.z;        // b*NH + h
    const int b = z >> 4, h = z & 15;
    float* C = scores + (long long)z * Ss * Ss;

    if (n0 > m0 + 31) {               // fully masked tile: overwrite poison
        #pragma unroll
        for (int mi = 0; mi < 2; ++mi)
          #pragma unroll
          for (int j = 0; j < 4; ++j)
            #pragma unroll
            for (int e = 0; e < 8; ++e) {
                const int r = m0 + mi * 16 + half * 8 + e;
                const int c = n0 + j * 16 + l15;
                C[(long long)r * Ss + c] = MASK_MIN;
            }
        return;
    }

    const __bf16* a0 = qcat + ((long long)(b * Ss + m0 + l15)) * (NHh * QKC) + h * QKC;
    const __bf16* a1 = a0 + 16LL * (NHh * QKC);
    const __bf16* bbase = kcat + ((long long)(b * Ss + n0 + l15)) * QKC;
    const __bf16* b0 = bbase;
    const __bf16* b1 = bbase + 16LL * QKC;
    const __bf16* b2 = bbase + 32LL * QKC;
    const __bf16* b3 = bbase + 48LL * QKC;

    f32x8 acc[2][4] = {};
    for (int kk = 0; kk < QKC; kk += 32) {
        bf16x16 af0 = load_a_frag(a0, kk, half);
        bf16x16 af1 = load_a_frag(a1, kk, half);
        bf16x16 bf0 = load_b_frag(b0, kk, half);
        bf16x16 bf1 = load_b_frag(b1, kk, half);
        bf16x16 bf2 = load_b_frag(b2, kk, half);
        bf16x16 bf3 = load_b_frag(b3, kk, half);
        acc[0][0] = wmma_bf16(af0, bf0, acc[0][0]);
        acc[0][1] = wmma_bf16(af0, bf1, acc[0][1]);
        acc[0][2] = wmma_bf16(af0, bf2, acc[0][2]);
        acc[0][3] = wmma_bf16(af0, bf3, acc[0][3]);
        acc[1][0] = wmma_bf16(af1, bf0, acc[1][0]);
        acc[1][1] = wmma_bf16(af1, bf1, acc[1][1]);
        acc[1][2] = wmma_bf16(af1, bf2, acc[1][2]);
        acc[1][3] = wmma_bf16(af1, bf3, acc[1][3]);
    }
    const float scale = 0.07216878364870323f; // 1/sqrt(192)
    #pragma unroll
    for (int mi = 0; mi < 2; ++mi)
      #pragma unroll
      for (int j = 0; j < 4; ++j)
        #pragma unroll
        for (int e = 0; e < 8; ++e) {
            const int r = m0 + mi * 16 + half * 8 + e;
            const int c = n0 + j * 16 + l15;
            const float v = (c <= r) ? acc[mi][j][e] * scale : MASK_MIN;
            C[(long long)r * Ss + c] = v;
        }
}

// ---------------- softmax over 1024-key rows, bf16 P written in-place over f32 scores ------
__global__ void softmax_rows(float* __restrict__ scores)
{
    const long long row = blockIdx.x;
    float* srow = scores + row * (long long)Ss;
    __bf16* prow = reinterpret_cast<__bf16*>(srow);   // row stride stays 4096 B (lda=2048 bf16)
    const int tid = threadIdx.x;
    float v[4];
    float mx = MASK_MIN;
    #pragma unroll
    for (int i = 0; i < 4; ++i) { v[i] = srow[tid + i * 256]; mx = fmaxf(mx, v[i]); }
    __shared__ float red[256];
    red[tid] = mx; __syncthreads();
    for (int k = 128; k > 0; k >>= 1) { if (tid < k) red[tid] = fmaxf(red[tid], red[tid + k]); __syncthreads(); }
    mx = red[0]; __syncthreads();
    float sum = 0.f;
    #pragma unroll
    for (int i = 0; i < 4; ++i) { v[i] = __expf(v[i] - mx); sum += v[i]; }
    red[tid] = sum; __syncthreads();
    for (int k = 128; k > 0; k >>= 1) { if (tid < k) red[tid] += red[tid + k]; __syncthreads(); }
    const float inv = 1.0f / red[0];
    #pragma unroll
    for (int i = 0; i < 4; ++i) prow[tid + i * 256] = f2bf(v[i] * inv);
}

// ---------------- LayerNorm rows (f32 in -> bf16 out) ----------------
__global__ void layernorm_rows(const float* __restrict__ x, const float* __restrict__ w,
                               __bf16* __restrict__ y, int D)
{
    const long long row = blockIdx.x;
    const float* xr = x + row * (long long)D;
    __bf16* yr = y + row * (long long)D;
    const int tid = threadIdx.x;
    float s = 0.f, s2 = 0.f;
    for (int i = tid; i < D; i += 256) { float t = xr[i]; s += t; s2 += t * t; }
    __shared__ float r1[256], r2[256];
    r1[tid] = s; r2[tid] = s2; __syncthreads();
    for (int k = 128; k > 0; k >>= 1) {
        if (tid < k) { r1[tid] += r1[tid + k]; r2[tid] += r2[tid + k]; }
        __syncthreads();
    }
    const float mean = r1[0] / D;
    const float var  = r2[0] / D - mean * mean;
    const float rs   = rsqrtf(var + 1e-5f);
    for (int i = tid; i < D; i += 256) yr[i] = f2bf((xr[i] - mean) * rs * w[i]);
}

// ---------------- kv_a post: LN(512)->kcat[:512] + kv_c^T ; rope(64)->kcat[512:576] -------
__global__ void kva_post(const float* __restrict__ kva, const float* __restrict__ w,
                         const float* __restrict__ cosb, const float* __restrict__ sinb,
                         __bf16* __restrict__ kcat, __bf16* __restrict__ kvcT)
{
    const long long row = blockIdx.x;        // b*S + s
    const int b = (int)(row / Ss);
    const int s = (int)(row % Ss);
    const float* xr = kva + row * QKC;
    const int tid = threadIdx.x;             // 256
    float sm = 0.f, s2 = 0.f;
    for (int i = tid; i < KVLRd; i += 256) { float t = xr[i]; sm += t; s2 += t * t; }
    __shared__ float r1[256], r2[256];
    r1[tid] = sm; r2[tid] = s2; __syncthreads();
    for (int k = 128; k > 0; k >>= 1) {
        if (tid < k) { r1[tid] += r1[tid + k]; r2[tid] += r2[tid + k]; }
        __syncthreads();
    }
    const float mean = r1[0] / KVLRd;
    const float var  = r2[0] / KVLRd - mean * mean;
    const float rs   = rsqrtf(var + 1e-5f);
    for (int i = tid; i < KVLRd; i += 256) {
        const __bf16 bv = f2bf((xr[i] - mean) * rs * w[i]);
        kcat[row * QKC + i] = bv;
        kvcT[((long long)b * KVLRd + i) * Ss + s] = bv;     // [B][512][S] for ctx GEMM B-operand
    }
    if (tid < ROPEd) {
        const float x = xr[KVLRd + tid];
        const float other = (tid < 32) ? -xr[KVLRd + tid + 32] : xr[KVLRd + tid - 32];
        const float r = x * cosb[row * ROPEd + tid] + other * sinb[row * ROPEd + tid];
        kcat[row * QKC + KVLRd + tid] = f2bf(r);
    }
}

// ---------------- q post: split q into q_nope (bf16) and roped q_pe -> qcat[512:576] ------
__global__ void q_post(const float* __restrict__ q, const float* __restrict__ cosb,
                       const float* __restrict__ sinb,
                       __bf16* __restrict__ q_nope, __bf16* __restrict__ qcat)
{
    const long long blk = blockIdx.x;
    const int h = (int)(blk % NHh);
    const long long row = blk / NHh;          // b*S + s
    const float* qr = q + (row * NHh + h) * QKDd;
    const int t = threadIdx.x;
    if (t < NOPEd) {
        q_nope[(row * NHh + h) * NOPEd + t] = f2bf(qr[t]);
    } else {
        const int i = t - NOPEd;
        const float x = qr[NOPEd + i];
        const float other = (i < 32) ? -qr[NOPEd + i + 32] : qr[NOPEd + i - 32];
        const float r = x * cosb[row * ROPEd + i] + other * sinb[row * ROPEd + i];
        qcat[(row * NHh + h) * QKC + KVLRd + i] = f2bf(r);
    }
}

// ---------------- casts ----------------
__global__ void cast_f32_bf16(const float* __restrict__ x, __bf16* __restrict__ y, long long n)
{
    for (long long i = blockIdx.x * (long long)blockDim.x + threadIdx.x; i < n;
         i += (long long)gridDim.x * blockDim.x)
        y[i] = f2bf(x[i]);
}
__global__ void transpose_cast(const float* __restrict__ x, __bf16* __restrict__ y,
                               int Z, int R, int C)
{
    const long long total = (long long)Z * R * C;
    for (long long i = blockIdx.x * (long long)blockDim.x + threadIdx.x; i < total;
         i += (long long)gridDim.x * blockDim.x) {
        const int c = (int)(i % C);
        const long long t = i / C;
        const int r = (int)(t % R);
        const int z = (int)(t / R);
        y[((long long)z * C + c) * R + r] = f2bf(x[i]);
    }
}

// ================= host side =================
extern "C" void kernel_launch(void* const* d_in, const int* in_sizes, int n_in,
                              void* d_out, int out_size, void* d_ws, size_t ws_size,
                              hipStream_t stream)
{
    const float* hidden = (const float*)d_in[0];
    const float* cosb   = (const float*)d_in[1];
    const float* sinb   = (const float*)d_in[2];
    const float* w_qa   = (const float*)d_in[3];
    const float* q_ln_w = (const float*)d_in[4];
    const float* w_qb   = (const float*)d_in[5];
    const float* w_kva  = (const float*)d_in[6];
    const float* kv_ln_w= (const float*)d_in[7];
    const float* w_ukt  = (const float*)d_in[8];
    const float* w_uv   = (const float*)d_in[9];
    const float* w_o    = (const float*)d_in[10];
    float* out = (float*)d_out;

    // ---- workspace carve-up (256B aligned) ----
    char* base = (char*)d_ws;
    size_t off = 0;
    auto alloc = [&](size_t bytes) -> char* {
        char* p = base + off;
        off = (off + bytes + 255) & ~(size_t)255;
        return p;
    };
    __bf16* hid_bf  = (__bf16*)alloc((size_t)Mrows * Hh * 2);
    __bf16* wqa_bf  = (__bf16*)alloc((size_t)QLRd * Hh * 2);
    __bf16* wqb_bf  = (__bf16*)alloc((size_t)NHh * QKDd * QLRd * 2);
    __bf16* wkva_bf = (__bf16*)alloc((size_t)QKC * Hh * 2);
    __bf16* wukt_t  = (__bf16*)alloc((size_t)NHh * KVLRd * NOPEd * 2); // [h][l][d]
    __bf16* wuv_t   = (__bf16*)alloc((size_t)NHh * VDd * KVLRd * 2);   // [h][v][l]
    __bf16* wo_bf   = (__bf16*)alloc((size_t)Hh * NHh * VDd * 2);
    float*  qa_raw  = (float*)alloc((size_t)Mrows * QLRd * 4);
    __bf16* qa_ln   = (__bf16*)alloc((size_t)Mrows * QLRd * 2);
    float*  q_f32   = (float*)alloc((size_t)Mrows * NHh * QKDd * 4);
    float*  kva_f32 = (float*)alloc((size_t)Mrows * QKC * 4);
    __bf16* q_nope  = (__bf16*)alloc((size_t)Mrows * NHh * NOPEd * 2);
    __bf16* qcat    = (__bf16*)alloc((size_t)Mrows * NHh * QKC * 2);
    __bf16* kcat    = (__bf16*)alloc((size_t)Mrows * QKC * 2);
    __bf16* kvcT    = (__bf16*)alloc((size_t)Bb * KVLRd * Ss * 2);
    float*  scores  = (float*)alloc((size_t)Bb * NHh * Ss * Ss * 4); // P bf16 reuses this
    __bf16* ctx     = (__bf16*)alloc((size_t)Bb * NHh * Ss * KVLRd * 2);
    __bf16* out1    = (__bf16*)alloc((size_t)Mrows * NHh * VDd * 2);
    (void)ws_size; (void)in_sizes; (void)n_in; (void)out_size;

    // ---- 1) casts / layout transforms ----
    cast_f32_bf16<<<4096, 256, 0, stream>>>(hidden, hid_bf, (long long)Mrows * Hh);
    cast_f32_bf16<<<4096, 256, 0, stream>>>(w_qa,  wqa_bf,  (long long)QLRd * Hh);
    cast_f32_bf16<<<4096, 256, 0, stream>>>(w_qb,  wqb_bf,  (long long)NHh * QKDd * QLRd);
    cast_f32_bf16<<<4096, 256, 0, stream>>>(w_kva, wkva_bf, (long long)QKC * Hh);
    cast_f32_bf16<<<4096, 256, 0, stream>>>(w_o,   wo_bf,   (long long)Hh * NHh * VDd);
    transpose_cast<<<4096, 256, 0, stream>>>(w_ukt, wukt_t, NHh, NOPEd, KVLRd); // ->[h][512][128]
    transpose_cast<<<4096, 256, 0, stream>>>(w_uv,  wuv_t,  NHh, KVLRd, VDd);   // ->[h][128][512]

    // ---- 2) qa = hidden @ w_qa^T  (f32 out) ----
    gemm_wmma_lds<false><<<dim3(QLRd/128, Mrows/64, 1), 128, 0, stream>>>(
        hid_bf, Hh, 0, wqa_bf, Hh, 0, qa_raw, QLRd, 0, Hh);
    layernorm_rows<<<Mrows, 256, 0, stream>>>(qa_raw, q_ln_w, qa_ln, QLRd);

    // ---- 3) q = qa_ln @ w_qb^T  (f32 out), then split/rope ----
    gemm_wmma_lds<false><<<dim3((NHh*QKDd)/128, Mrows/64, 1), 128, 0, stream>>>(
        qa_ln, QLRd, 0, wqb_bf, QLRd, 0, q_f32, NHh*QKDd, 0, QLRd);

    // ---- 4) kva = hidden @ w_kva^T (N=576 not /128 -> wave kernel), then LN + rope ----
    gemm_wmma<false><<<dim3(QKC/64, Mrows/32, 1), 32, 0, stream>>>(
        hid_bf, Hh, 0, wkva_bf, Hh, 0, kva_f32, QKC, 0, Hh);
    kva_post<<<Mrows, 256, 0, stream>>>(kva_f32, kv_ln_w, cosb, sinb, kcat, kvcT);
    q_post<<<Mrows * NHh, 192, 0, stream>>>(q_f32, cosb, sinb, q_nope, qcat);

    // ---- 5) q_lat = q_nope @ W_UK_T  (per head, bf16 out -> qcat[:,:,0:512]) ----
    gemm_wmma_lds<true><<<dim3(KVLRd/128, Mrows/64, NHh), 128, 0, stream>>>(
        q_nope, NHh*NOPEd, NOPEd,
        wukt_t, NOPEd, (long long)KVLRd*NOPEd,
        qcat, NHh*QKC, QKC, NOPEd);

    // ---- 6) scores = qcat @ kcat^T (causal, scaled) ; softmax -> P (bf16 in-place) ----
    attn_scores<<<dim3(Ss/64, Ss/32, Bb*NHh), 32, 0, stream>>>(qcat, kcat, scores);
    softmax_rows<<<Bb*NHh*Ss, 256, 0, stream>>>(scores);

    // ---- 7) ctx = P @ kv_c  (per b: heads in grid.z), P row stride = 2048 bf16 ----
    for (int b = 0; b < Bb; ++b) {
        gemm_wmma_lds<true><<<dim3(KVLRd/128, Ss/64, NHh), 128, 0, stream>>>(
            (const __bf16*)scores + (long long)b*NHh*Ss*(2*Ss), 2*Ss, (long long)Ss*(2*Ss),
            kvcT + (long long)b*KVLRd*Ss, Ss, 0,
            ctx + (long long)b*NHh*Ss*KVLRd, KVLRd, (long long)Ss*KVLRd, Ss);
    }
    // ---- 8) out1[b,q,h*128+v] = ctx @ W_UV (per head) ----
    for (int b = 0; b < Bb; ++b) {
        gemm_wmma_lds<true><<<dim3(VDd/128, Ss/64, NHh), 128, 0, stream>>>(
            ctx + (long long)b*NHh*Ss*KVLRd, KVLRd, (long long)Ss*KVLRd,
            wuv_t, KVLRd, (long long)VDd*KVLRd,
            out1 + (long long)b*Ss*NHh*VDd, NHh*VDd, VDd, KVLRd);
    }
    // ---- 9) out = out1 @ w_o^T  (f32) ----
    gemm_wmma_lds<false><<<dim3(Hh/128, Mrows/64, 1), 128, 0, stream>>>(
        out1, NHh*VDd, 0, wo_bf, NHh*VDd, 0, out, Hh, 0, NHh*VDd);
}
